// Net_40596030881968
// MI455X (gfx1250) — compile-verified, matrix operand order checked by reading, not run on hardware
//
#include <hip/hip_runtime.h>
#include <cmath>

typedef __attribute__((ext_vector_type(16))) _Float16 v16h;
typedef __attribute__((ext_vector_type(8)))  float    v8f;

#define EMB     64
#define NGRAPH  512

// Hardware f32 atomic add (global_atomic_add_f32) — avoids CAS loops on the
// L2-resident scatter path, which is the throughput limiter for this workload.
__device__ inline void atomAddF(float* p, float v) { unsafeAtomicAdd(p, v); }

// ---------------- degree / norm ----------------
__global__ void k_deg_init(float* __restrict__ deg, int n) {
  int i = blockIdx.x * blockDim.x + threadIdx.x;
  if (i < n) deg[i] = 1.0f;  // self-loop contributes 1 to every node's degree
}

__global__ void k_deg_acc(const long long* __restrict__ dst, float* __restrict__ deg, int E) {
  int e = blockIdx.x * blockDim.x + threadIdx.x;
  if (e < E) atomAddF(&deg[(int)dst[e]], 1.0f);
}

__global__ void k_dinv(float* __restrict__ d, int n) {
  int i = blockIdx.x * blockDim.x + threadIdx.x;
  if (i < n) d[i] = rsqrtf(d[i]);  // deg >= 1 always (self-loops)
}

__global__ void k_zero(float* __restrict__ p, int n) {
  int i = blockIdx.x * blockDim.x + threadIdx.x;
  if (i < n) p[i] = 0.0f;
}

// ---------------- layer-0 GEMM: [N,9] @ [9,64]  (K too small for WMMA) -----
__global__ void k_gemm_in(const float* __restrict__ x, const float* __restrict__ W,
                          float* __restrict__ T, int n) {
  int i = blockIdx.x * blockDim.x + threadIdx.x;
  if (i >= n * EMB) return;
  int node = i >> 6, o = i & 63;
  const float* xp = x + node * 9;
  float s = 0.f;
#pragma unroll
  for (int k = 0; k < 9; ++k) s = fmaf(xp[k], W[k * EMB + o], s);
  T[i] = s;
}

// ---------------- WMMA GEMM: [N,64] @ [64,64] ------------------------------
// One wave computes a 16x16 tile of T; 4 waves/block cover the 64 columns.
// f16 inputs, f32 accumulate (v_wmma_f32_16x16x32_f16), K=64 -> 2 WMMAs.
__global__ void k_gemm64_wmma(const float* __restrict__ Hin, const float* __restrict__ W,
                              float* __restrict__ T, int n) {
  const int lane = threadIdx.x & 31;
  const int wv   = threadIdx.x >> 5;      // 0..3 : column tile
  const int half = lane >> 4;             // 0 or 1
  const int l16  = lane & 15;
  const int mbase = blockIdx.x * 16;
  int arow = mbase + l16;                 // A-matrix row for this lane
  if (arow >= n) arow = n - 1;            // clamp (keeps EXEC all-ones for WMMA)
  const int col = wv * 16 + l16;          // B/D column for this lane

  v8f c = {};
#pragma unroll
  for (int k0 = 0; k0 < EMB; k0 += 32) {
    v16h a, b;
#pragma unroll
    for (int j = 0; j < 16; ++j) {
      // 16-bit A 16x32 layout: lane half picks K block; j>=8 jumps to K+16
      const int k = k0 + j + 8 * half + ((j >> 3) << 3);
      a[j] = (_Float16)Hin[arow * EMB + k];
      b[j] = (_Float16)W[k * EMB + col];
    }
    c = __builtin_amdgcn_wmma_f32_16x16x32_f16(false, a, false, b,
                                               (short)0, c, false, false);
  }

  // Uniform-branch epilogue: all-full tile (the only case when n % 16 == 0)
  // takes the straight-line 8-store path with no exec juggling.
  if (mbase + 16 <= n) {
#pragma unroll
    for (int r = 0; r < 8; ++r)
      T[(mbase + r + 8 * half) * EMB + col] = c[r];
  } else {
#pragma unroll
    for (int r = 0; r < 8; ++r) {
      const int m = mbase + r + 8 * half;
      if (m < n) T[m * EMB + col] = c[r];
    }
  }
}

// ---------------- edge aggregation: AGG[dst] += T[src] * dinv[s]*dinv[d] ---
// 32 threads per edge, 2 features each (float2 gather -> b64 loads, halves
// the redundant index traffic). Scatter via native f32 L2 atomics.
__global__ void k_aggregate(const float* __restrict__ T,
                            const long long* __restrict__ src,
                            const long long* __restrict__ dst,
                            const float* __restrict__ dinv,
                            float* __restrict__ agg, int E) {
  int t = blockIdx.x * blockDim.x + threadIdx.x;
  int e = t >> 5;
  if (e >= E) return;
  int f = (t & 31) * 2;
  int s = (int)src[e];
  int d = (int)dst[e];
  float nrm = dinv[s] * dinv[d];
  float2 m = *(const float2*)(T + s * EMB + f);
  atomAddF(&agg[d * EMB + f],     m.x * nrm);
  atomAddF(&agg[d * EMB + f + 1], m.y * nrm);
}

// ------- epilogue: fold self-loop message, add bias, tanh ------------------
__global__ void k_finalize(const float* __restrict__ agg, const float* __restrict__ T,
                           const float* __restrict__ dinv, const float* __restrict__ b,
                           float* __restrict__ hout, int n) {
  int i = blockIdx.x * blockDim.x + threadIdx.x;
  if (i >= n * EMB) return;
  int node = i >> 6, f = i & 63;
  float di = dinv[node];
  hout[i] = tanhf(agg[i] + T[i] * di * di + b[f]);
}

// ---------------- pooling ---------------------------------------------------
__device__ inline void atomicMaxF(float* addr, float v) {
  if (v >= 0.f) atomicMax((int*)addr, __float_as_int(v));
  else          atomicMin((unsigned int*)addr, __float_as_uint(v));
}

__global__ void k_pool_init(float* __restrict__ gmax, float* __restrict__ gsum,
                            float* __restrict__ cnt, int nG) {
  int i = blockIdx.x * blockDim.x + threadIdx.x;
  if (i < nG * EMB) { gmax[i] = -INFINITY; gsum[i] = 0.f; }
  if (i < nG) cnt[i] = 0.f;
}

__global__ void k_pool(const float* __restrict__ h, const long long* __restrict__ batch,
                       float* __restrict__ gmax, float* __restrict__ gsum,
                       float* __restrict__ cnt, int n) {
  int i = blockIdx.x * blockDim.x + threadIdx.x;
  if (i >= n * EMB) return;
  int node = i >> 6, f = i & 63;
  int g = (int)batch[node];
  float v = h[i];
  atomicMaxF(&gmax[g * EMB + f], v);
  atomAddF(&gsum[g * EMB + f], v);
  if (f == 0) atomAddF(&cnt[g], 1.0f);
}

// ---------------- head: hidden = [gmax | gmean], out = hidden@Wout + bout --
__global__ void k_head(const float* __restrict__ gmax, const float* __restrict__ gsum,
                       const float* __restrict__ cnt, const float* __restrict__ Wout,
                       const float* __restrict__ bout, float* __restrict__ out, int nG) {
  int g = blockIdx.x;
  int f = threadIdx.x;            // 0..127
  float hv;
  if (f < EMB) {
    float m = gmax[g * EMB + f];
    hv = isfinite(m) ? m : 0.0f;  // empty-graph guard
  } else {
    float c = cnt[g];
    hv = gsum[g * EMB + (f - EMB)] / fmaxf(c, 1.0f);
  }
  out[nG + g * (2 * EMB) + f] = hv;   // hidden block follows out block

  __shared__ float red[2 * EMB];
  red[f] = hv * Wout[f];              // Wout: [128,1]
  __syncthreads();
  for (int s = EMB; s > 0; s >>= 1) {
    if (f < s) red[f] += red[f + s];
    __syncthreads();
  }
  if (f == 0) out[g] = red[0] + bout[0];
}

// ---------------------------------------------------------------------------
extern "C" void kernel_launch(void* const* d_in, const int* in_sizes, int n_in,
                              void* d_out, int out_size, void* d_ws, size_t ws_size,
                              hipStream_t stream) {
  const float*     x     = (const float*)d_in[0];
  const long long* ei    = (const long long*)d_in[1];   // int64 [2, E]
  const long long* batch = (const long long*)d_in[2];   // int64 [N]
  const float* W0   = (const float*)d_in[3];
  const float* b0   = (const float*)d_in[4];
  const float* W1   = (const float*)d_in[5];
  const float* b1   = (const float*)d_in[6];
  const float* W2   = (const float*)d_in[7];
  const float* b2   = (const float*)d_in[8];
  const float* Wout = (const float*)d_in[9];
  const float* bout = (const float*)d_in[10];

  const int N = in_sizes[0] / 9;
  const int E = in_sizes[1] / 2;
  const long long* src = ei;
  const long long* dst = ei + E;

  float* ws   = (float*)d_ws;
  float* dinv = ws;  ws += N;
  float* T    = ws;  ws += (size_t)N * EMB;
  float* AGG  = ws;  ws += (size_t)N * EMB;
  float* H    = ws;  ws += (size_t)N * EMB;
  float* gmax = ws;  ws += NGRAPH * EMB;
  float* gsum = ws;  ws += NGRAPH * EMB;
  float* cnt  = ws;  ws += NGRAPH;

  auto cdiv = [](int a, int b) { return (a + b - 1) / b; };
  const int NE = N * EMB;        // 6.4M
  const int E2 = E * 32;         // threads for aggregate (2 feats/thread)

  // symmetric normalization
  k_deg_init<<<cdiv(N, 256), 256, 0, stream>>>(dinv, N);
  k_deg_acc <<<cdiv(E, 256), 256, 0, stream>>>(dst, dinv, E);
  k_dinv    <<<cdiv(N, 256), 256, 0, stream>>>(dinv, N);

  const float* biases[3] = { b0, b1, b2 };
  for (int layer = 0; layer < 3; ++layer) {
    if (layer == 0)
      k_gemm_in<<<cdiv(NE, 256), 256, 0, stream>>>(x, W0, T, N);
    else
      k_gemm64_wmma<<<cdiv(N, 16), 128, 0, stream>>>(H, layer == 1 ? W1 : W2, T, N);

    k_zero     <<<cdiv(NE, 256), 256, 0, stream>>>(AGG, NE);
    k_aggregate<<<cdiv(E2, 256), 256, 0, stream>>>(T, src, dst, dinv, AGG, E);
    k_finalize <<<cdiv(NE, 256), 256, 0, stream>>>(AGG, T, dinv, biases[layer], H, N);
  }

  k_pool_init<<<cdiv(NGRAPH * EMB, 256), 256, 0, stream>>>(gmax, gsum, cnt, NGRAPH);
  k_pool     <<<cdiv(NE, 256), 256, 0, stream>>>(H, batch, gmax, gsum, cnt, N);
  k_head     <<<NGRAPH, 2 * EMB, 0, stream>>>(gmax, gsum, cnt, Wout, bout,
                                              (float*)d_out, NGRAPH);
}
